// paperLSTM_62062277427267
// MI455X (gfx1250) — matrix operand order repaired and checked by reading, not compiled, not run on hardware
//
#include <hip/hip_runtime.h>

#define TT 512
#define BB 128
#define II 512
#define HH 1024
#define HI 1536   /* H + I */
#define NBLK 128  /* (HH/16) * (BB/64) persistent workgroups */

typedef __attribute__((ext_vector_type(16))) __bf16 v16bf;
typedef __attribute__((ext_vector_type(8)))  __bf16 v8bf;
typedef __attribute__((ext_vector_type(8)))  float  v8f;

__device__ __forceinline__ float fsig(float x) {
    return 1.0f / (1.0f + __expf(-x));
}
__device__ __forceinline__ float ftanh(float x) {
    float ax = fabsf(x);
    float e  = __expf(-2.0f * ax);
    float r  = (1.0f - e) / (1.0f + e);
    return x < 0.0f ? -r : r;
}

// ---- conversion / packing / init kernels ----------------------------------

__global__ void cvt_bf16_kernel(const float* __restrict__ src,
                                __bf16* __restrict__ dst, long n) {
    long i = (long)blockIdx.x * blockDim.x + threadIdx.x;
    long stride = (long)gridDim.x * blockDim.x;
    for (; i < n; i += stride) dst[i] = (__bf16)src[i];
}

__global__ void pack_w_kernel(const float* __restrict__ Wf,
                              const float* __restrict__ Wi,
                              const float* __restrict__ Wc,
                              const float* __restrict__ Wo,
                              __bf16* __restrict__ dst) {
    const long n = (long)HH * HI;
    long i = (long)blockIdx.x * blockDim.x + threadIdx.x;
    long stride = (long)gridDim.x * blockDim.x;
    for (; i < 4 * n; i += stride) {
        int g = (int)(i / n);
        long r = i % n;
        const float* s = (g == 0) ? Wf : (g == 1) ? Wi : (g == 2) ? Wc : Wo;
        dst[i] = (__bf16)s[r];
    }
}

// zero h0, c0, and the per-step barrier counters (replay-deterministic)
__global__ void init_state_kernel(__bf16* __restrict__ h, float* __restrict__ c,
                                  unsigned* __restrict__ bar, int n) {
    int i = blockIdx.x * blockDim.x + threadIdx.x;
    if (i < n) { h[i] = (__bf16)0.0f; c[i] = 0.0f; }
    if (i < TT) bar[i] = 0u;
}

// ---- persistent LSTM kernel ------------------------------------------------
// grid = (HH/16, BB/64) = 128 resident workgroups, 256 threads (8 waves).
// Block owns slab rows [mbase,mbase+64) x cols [n0,n0+16) for all 4 gates,
// loops over all T steps, device-barrier between steps (only h crosses blocks).

__global__ __launch_bounds__(256) void lstm_persistent_kernel(
    const __bf16* __restrict__ xb,    // (TT, BB, II) bf16
    const __bf16* __restrict__ Wpk,   // (4, HH, HI) bf16 packed [f,i,c,o]
    const float*  __restrict__ bfp, const float* __restrict__ bip,
    const float*  __restrict__ bcp, const float* __restrict__ bop,
    __bf16*       __restrict__ h0buf, // (BB, HH) bf16 ping
    __bf16*       __restrict__ h1buf, // (BB, HH) bf16 pong
    float*        __restrict__ c,     // (BB, HH) f32, in place
    float*        __restrict__ out,   // (TT, BB, HH) f32
    unsigned*     __restrict__ bar)   // TT arrival counters, pre-zeroed
{
    __shared__ float pre[4][64][16];

    const int tid   = threadIdx.x;
    const int lane  = tid & 31;
    const int wave  = tid >> 5;
    const int g     = wave & 3;
    const int mp    = wave >> 2;           // 0..1 (32-row half within slab)
    const int n0    = blockIdx.x << 4;     // 16-column slab in H
    const int mbase = blockIdx.y << 6;     // 64-row slab in B

    const int lrow = lane & 15;
    const int kq   = lane >> 4;            // 0/1 lane half
    const int ka8  = kq << 3;              // A-fragment K sub-offset (+0/+8)
    const int kb16 = kq << 4;              // unused (+0/+16); B loads also use ka8

    const int r0 = mbase + (mp << 5) + lrow;   // row of A tile 0 (this lane)
    const int r1 = r0 + 16;                    // row of A tile 1

    const size_t hoff0 = (size_t)r0 * HH;
    const size_t hoff1 = (size_t)r1 * HH;
    const size_t xoff0 = (size_t)r0 * II;
    const size_t xoff1 = (size_t)r1 * II;
    const __bf16* wrow = Wpk + ((size_t)g * HH + (size_t)(n0 + lrow)) * HI;

    for (int t = 0; t < TT; ++t) {
        const __bf16* hin  = (t & 1) ? h1buf : h0buf;
        __bf16*       hout = (t & 1) ? h0buf : h1buf;
        const __bf16* xt   = xb + (size_t)t * BB * II;
        float*        outt = out + (size_t)t * BB * HH;

        const __bf16* ah0 = hin + hoff0;
        const __bf16* ah1 = hin + hoff1;
        const __bf16* ax0 = xt + xoff0;
        const __bf16* ax1 = xt + xoff1;

        v8f acc0 = {};
        v8f acc1 = {};

        // K over recurrent part: z[:, 0:1024) = h_prev
        for (int k0 = 0; k0 < HH; k0 += 32) {
            v16bf a0, a1;
            {
                v8bf lo = *(const v8bf*)(ah0 + k0 + ka8);
                v8bf hi = *(const v8bf*)(ah0 + k0 + ka8 + 16);
#pragma unroll
                for (int i = 0; i < 8; ++i) { a0[i] = lo[i]; a0[i + 8] = hi[i]; }
            }
            {
                v8bf lo = *(const v8bf*)(ah1 + k0 + ka8);
                v8bf hi = *(const v8bf*)(ah1 + k0 + ka8 + 16);
#pragma unroll
                for (int i = 0; i < 8; ++i) { a1[i] = lo[i]; a1[i + 8] = hi[i]; }
            }
            v16bf b;
            {
                v8bf lo = *(const v8bf*)(wrow + k0 + ka8);
                v8bf hi = *(const v8bf*)(wrow + k0 + ka8 + 16);
#pragma unroll
                for (int i = 0; i < 8; ++i) { b[i] = lo[i]; b[i + 8] = hi[i]; }
            }
            acc0 = __builtin_amdgcn_wmma_f32_16x16x32_bf16(false, a0, false, b,
                                                           (short)0, acc0, false, false);
            acc1 = __builtin_amdgcn_wmma_f32_16x16x32_bf16(false, a1, false, b,
                                                           (short)0, acc1, false, false);
        }

        // K over input part: z[:, 1024:1536) = x_t
        for (int k0 = 0; k0 < II; k0 += 32) {
            v16bf a0, a1;
            {
                v8bf lo = *(const v8bf*)(ax0 + k0 + ka8);
                v8bf hi = *(const v8bf*)(ax0 + k0 + ka8 + 16);
#pragma unroll
                for (int i = 0; i < 8; ++i) { a0[i] = lo[i]; a0[i + 8] = hi[i]; }
            }
            {
                v8bf lo = *(const v8bf*)(ax1 + k0 + ka8);
                v8bf hi = *(const v8bf*)(ax1 + k0 + ka8 + 16);
#pragma unroll
                for (int i = 0; i < 8; ++i) { a1[i] = lo[i]; a1[i + 8] = hi[i]; }
            }
            v16bf b;
            {
                v8bf lo = *(const v8bf*)(wrow + HH + k0 + ka8);
                v8bf hi = *(const v8bf*)(wrow + HH + k0 + ka8 + 16);
#pragma unroll
                for (int i = 0; i < 8; ++i) { b[i] = lo[i]; b[i + 8] = hi[i]; }
            }
            acc0 = __builtin_amdgcn_wmma_f32_16x16x32_bf16(false, a0, false, b,
                                                           (short)0, acc0, false, false);
            acc1 = __builtin_amdgcn_wmma_f32_16x16x32_bf16(false, a1, false, b,
                                                           (short)0, acc1, false, false);
        }

        // Spill accumulators: D layout is VGPR r -> M = tile + r + (lane>=16 ? 8 : 0)
        {
            const int n = lrow;
            const int mloc0 = (mp << 5) + (kq << 3);
#pragma unroll
            for (int r = 0; r < 8; ++r) pre[g][mloc0 + r][n] = acc0[r];
#pragma unroll
            for (int r = 0; r < 8; ++r) pre[g][mloc0 + 16 + r][n] = acc1[r];
        }
        __syncthreads();

        // Elementwise gate math for the 64x16 slab (1024 elems / 256 threads)
#pragma unroll
        for (int j = 0; j < 4; ++j) {
            int e = tid + j * 256;
            int m = e >> 4, n = e & 15;
            int grow = mbase + m;
            int gcol = n0 + n;
            float pf = pre[0][m][n] + bfp[gcol];
            float pi = pre[1][m][n] + bip[gcol];
            float pc = pre[2][m][n] + bcp[gcol];
            float po = pre[3][m][n] + bop[gcol];
            float fg = fsig(pf);
            float ig = fsig(pi);
            float ct = ftanh(pc);
            float og = fsig(po);
            size_t idx = (size_t)grow * HH + gcol;
            float cn = fg * c[idx] + ig * ct;
            c[idx] = cn;
            float hv = og * ftanh(cn);
            hout[idx] = (__bf16)hv;
            outt[idx] = hv;
        }

        // ---- device-wide step barrier (release h writes, acquire for reads)
        __syncthreads();      // all waves done writing hout + reading pre
        __threadfence();      // flush this wave's h stores to device scope
        if (tid == 0) {
            __hip_atomic_fetch_add(&bar[t], 1u, __ATOMIC_RELEASE,
                                   __HIP_MEMORY_SCOPE_AGENT);
            while (__hip_atomic_load(&bar[t], __ATOMIC_ACQUIRE,
                                     __HIP_MEMORY_SCOPE_AGENT) < (unsigned)NBLK) {
                __builtin_amdgcn_s_sleep(1);
            }
        }
        __syncthreads();      // whole block waits on tid0's acquire
    }
}

// ---- host launch -----------------------------------------------------------

extern "C" void kernel_launch(void* const* d_in, const int* in_sizes, int n_in,
                              void* d_out, int out_size, void* d_ws, size_t ws_size,
                              hipStream_t stream) {
    const float* x  = (const float*)d_in[0];
    const float* Wf = (const float*)d_in[1];
    const float* bf = (const float*)d_in[2];
    const float* Wi = (const float*)d_in[3];
    const float* bi = (const float*)d_in[4];
    const float* Wc = (const float*)d_in[5];
    const float* bc = (const float*)d_in[6];
    const float* Wo = (const float*)d_in[7];
    const float* bo = (const float*)d_in[8];
    float* out = (float*)d_out;

    char* ws = (char*)d_ws;
    size_t off = 0;
    auto take = [&](size_t bytes) -> char* {
        char* p = ws + off;
        off += (bytes + 255) & ~(size_t)255;
        return p;
    };

    __bf16*   xb  = (__bf16*)take((size_t)TT * BB * II * 2); // 67 MB
    __bf16*   Wpk = (__bf16*)take((size_t)4 * HH * HI * 2);  // 12.6 MB
    __bf16*   h0  = (__bf16*)take((size_t)BB * HH * 2);
    __bf16*   h1  = (__bf16*)take((size_t)BB * HH * 2);
    float*    cbf = (float*)take((size_t)BB * HH * 4);
    unsigned* bar = (unsigned*)take((size_t)TT * 4);
    (void)ws_size; (void)in_sizes; (void)n_in; (void)out_size;

    // Per-launch precision conversion + state/barrier init, all on stream.
    cvt_bf16_kernel<<<4096, 256, 0, stream>>>(x, xb, (long)TT * BB * II);
    pack_w_kernel<<<4096, 256, 0, stream>>>(Wf, Wi, Wc, Wo, Wpk);
    init_state_kernel<<<(BB * HH + 255) / 256, 256, 0, stream>>>(h0, cbf, bar, BB * HH);

    // Single persistent kernel runs all T=512 steps with internal grid sync.
    lstm_persistent_kernel<<<dim3(HH / 16, BB / 64), 256, 0, stream>>>(
        xb, Wpk, bf, bi, bc, bo, h0, h1, cbf, out, bar);
}
